// PiKVMoERouter_87308095193745
// MI455X (gfx1250) — matrix operand order repaired and checked by reading, not compiled, not run on hardware
//
#include <hip/hip_runtime.h>
#include <hip/hip_bf16.h>
#include <math.h>

// ---------------- problem constants ----------------
#define B_    2
#define S_    2048
#define H_    1024
#define E_    8
#define KTOP  2
#define MTOT  (B_ * S_)          // 4096 tokens
#define NSLOT (MTOT * KTOP)      // 8192 (b,s,k) slots
#define CAP   1536               // int(B*S*1.5*K/E)
#define DISPSZ ((size_t)MTOT * E_ * CAP)   // 50,331,648 floats per tensor

// ---------------- vector types for WMMA ----------------
typedef __attribute__((ext_vector_type(16))) __bf16 v16bf;
typedef __attribute__((ext_vector_type(8)))  __bf16 v8bf;
typedef __attribute__((ext_vector_type(8)))  float  v8f;

// ---------------- workspace layout (bytes) ----------------
#define OFF_XHI    ((size_t)0)                       // 4096*1024 bf16 = 8 MB
#define OFF_XLO    ((size_t)(8u  << 20))             // 8 MB
#define OFF_W1THI  ((size_t)(16u << 20))             // [N][K] bf16 = 2 MB
#define OFF_W1TLO  ((size_t)(18u << 20))             // 2 MB
#define OFF_H1     ((size_t)(20u << 20))             // 4096*1024 f32 = 16 MB
#define OFF_LOGITS ((size_t)(36u << 20))             // 4096*8 f32 = 128 KB
#define OFF_AVG    (OFF_LOGITS + (size_t)MTOT*E_*4)  // 2*1024 f32
#define OFF_ADJ    (OFF_AVG + (size_t)B_*H_*4)       // 2*8 f32
#define OFF_EFLAT  (OFF_ADJ + 256)                   // 8192 int
#define OFF_PFLAT  (OFF_EFLAT + (size_t)NSLOT*4)     // 8192 f32
#define OFF_PSUM   (OFF_PFLAT + (size_t)NSLOT*4)     // 8 f32
#define OFF_CNT    (OFF_PSUM + 64)                   // 8 u32

// =====================================================================
// bf16 split helpers
// =====================================================================
__device__ __forceinline__ void split2(float v, __bf16& hi, __bf16& lo) {
  __bf16 h = (__bf16)v;
  hi = h;
  lo = (__bf16)(v - (float)h);
}

// X [4096,1024] f32 -> hi/lo bf16 (same layout)
__global__ __launch_bounds__(256)
void split_x_kernel(const float* __restrict__ x, __bf16* __restrict__ hi,
                    __bf16* __restrict__ lo, int n) {
  int i = blockIdx.x * 256 + threadIdx.x;
  if (i < n) split2(x[i], hi[i], lo[i]);
}

// W1 [K=1024][N=1024] f32 -> transposed [N][K] hi/lo bf16
__global__ __launch_bounds__(256)
void split_w1t_kernel(const float* __restrict__ w, __bf16* __restrict__ thi,
                      __bf16* __restrict__ tlo) {
  int i = blockIdx.x * 256 + threadIdx.x;
  if (i < H_ * H_) {
    int k = i / H_, n = i - k * H_;
    __bf16 h, l;
    split2(w[i], h, l);
    thi[(size_t)n * H_ + k] = h;
    tlo[(size_t)n * H_ + k] = l;
  }
}

// =====================================================================
// avg_features: mean over S -> avg[B][H]
// =====================================================================
__global__ __launch_bounds__(256)
void avg_kernel(const float* __restrict__ hs, float* __restrict__ avg) {
  int h = blockIdx.x * 256 + threadIdx.x;
  int b = blockIdx.y;
  if (h < H_) {
    const float* p = hs + (size_t)b * S_ * H_ + h;
    float s = 0.f;
    for (int ss = 0; ss < S_; ++ss) s += p[(size_t)ss * H_];
    avg[b * H_ + h] = s * (1.0f / S_);
  }
}

// =====================================================================
// cache-aware adjustment: adj[b][e] = tanh(relu(combined@wc1+bc1)@wc2+bc2)
// =====================================================================
__global__ __launch_bounds__(256)
void cacheadj_kernel(const float* __restrict__ avg, const float* __restrict__ chr,
                     const float* __restrict__ wc1, const float* __restrict__ bc1,
                     const float* __restrict__ wc2, const float* __restrict__ bc2,
                     float* __restrict__ adj) {
  __shared__ float hc[H_];
  int b = blockIdx.x, tid = threadIdx.x;
  for (int j = tid; j < H_; j += 256) {
    float s = bc1[j];
    for (int i = 0; i < H_; ++i)  s += avg[b * H_ + i] * wc1[(size_t)i * H_ + j];
    for (int i = 0; i < E_; ++i)  s += chr[i] * wc1[(size_t)(H_ + i) * H_ + j];
    hc[j] = fmaxf(s, 0.f);
  }
  __syncthreads();
  if (tid < E_) {
    float s = bc2[tid];
    for (int j = 0; j < H_; ++j) s += hc[j] * wc2[(size_t)j * E_ + tid];
    adj[b * E_ + tid] = tanhf(s);
  }
}

// =====================================================================
// GEMM1 (WMMA): H1 = relu(X @ W1 + b1), split-bf16 3-term accumulation.
// Workgroup = 128 threads (4 waves). WG tile = 128(M) x 64(N).
// Each wave owns TWO 16-row M-subtiles (32 rows) x four 16-col N-subtiles,
// so every B fragment loaded from W1^T is used by 6 WMMAs (2 M-subtiles
// x 3 split terms) -> 24 WMMAs per 24 b128 loads per k-step per wave
// (vs 12:20 with one M-subtile). This relieves the VMEM port, which is
// the limiter given a b128 wave-load moves 512B vs a few-cycle WMMA.
//
// A frag (16x32 bf16, ISA layout): lane L (M = msub+L%16, half = L/16):
//   elems 0..7  = K k0 + 8*half + 0..7      (16B contiguous)
//   elems 8..15 = K k0 + 16 + 8*half + 0..7 (16B contiguous)
// B frag (32x16 bf16) from W1^T [N][K]: lane L (N = n0+L%16):
//   elems 0..7  = K k0 + 8*half + 0..7, elems 8..15 = K k0 + 16 + 8*half + 0..7
// C/D: elem r -> row msub+r+8*half, col n0+L%16.
// =====================================================================
__global__ __launch_bounds__(128)
void gemm1_wmma_kernel(const __bf16* __restrict__ xhi, const __bf16* __restrict__ xlo,
                       const __bf16* __restrict__ wthi, const __bf16* __restrict__ wtlo,
                       const float* __restrict__ b1, float* __restrict__ h1) {
  const int lane = threadIdx.x & 31;
  const int wave = threadIdx.x >> 5;         // 0..3
  const int half = lane >> 4;
  const int l16  = lane & 15;
  const int m0    = blockIdx.x * 128 + wave * 32;  // two subtiles: m0, m0+16
  const int nbase = blockIdx.y * 64;

  const __bf16* pAhi0 = xhi + (size_t)(m0 + l16) * H_;
  const __bf16* pAlo0 = xlo + (size_t)(m0 + l16) * H_;
  const __bf16* pAhi1 = xhi + (size_t)(m0 + 16 + l16) * H_;
  const __bf16* pAlo1 = xlo + (size_t)(m0 + 16 + l16) * H_;

  v8f acc0[4] = {v8f{}, v8f{}, v8f{}, v8f{}};   // M-subtile 0
  v8f acc1[4] = {v8f{}, v8f{}, v8f{}, v8f{}};   // M-subtile 1

  for (int k0 = 0; k0 < H_; k0 += 32) {
    // hint upcoming A chunks into cache
    __builtin_prefetch(pAhi0 + k0 + 64, 0, 1);
    __builtin_prefetch(pAlo0 + k0 + 64, 0, 1);
    __builtin_prefetch(pAhi1 + k0 + 64, 0, 1);
    __builtin_prefetch(pAlo1 + k0 + 64, 0, 1);

    const int ka = k0 + 8 * half;
    v8bf t0, t1;

    t0 = *(const v8bf*)(pAhi0 + ka);
    t1 = *(const v8bf*)(pAhi0 + ka + 16);
    v16bf Ahi0 = __builtin_shufflevector(t0, t1, 0,1,2,3,4,5,6,7,8,9,10,11,12,13,14,15);
    t0 = *(const v8bf*)(pAlo0 + ka);
    t1 = *(const v8bf*)(pAlo0 + ka + 16);
    v16bf Alo0 = __builtin_shufflevector(t0, t1, 0,1,2,3,4,5,6,7,8,9,10,11,12,13,14,15);
    t0 = *(const v8bf*)(pAhi1 + ka);
    t1 = *(const v8bf*)(pAhi1 + ka + 16);
    v16bf Ahi1 = __builtin_shufflevector(t0, t1, 0,1,2,3,4,5,6,7,8,9,10,11,12,13,14,15);
    t0 = *(const v8bf*)(pAlo1 + ka);
    t1 = *(const v8bf*)(pAlo1 + ka + 16);
    v16bf Alo1 = __builtin_shufflevector(t0, t1, 0,1,2,3,4,5,6,7,8,9,10,11,12,13,14,15);

#pragma unroll
    for (int nt = 0; nt < 4; ++nt) {
      const int col = nbase + 16 * nt + l16;
      const __bf16* pBh = wthi + (size_t)col * H_ + k0 + 8 * half;
      const __bf16* pBl = wtlo + (size_t)col * H_ + k0 + 8 * half;
      v8bf u0 = *(const v8bf*)(pBh);
      v8bf u1 = *(const v8bf*)(pBh + 16);
      v16bf Bhi = __builtin_shufflevector(u0, u1, 0,1,2,3,4,5,6,7,8,9,10,11,12,13,14,15);
      v8bf u2 = *(const v8bf*)(pBl);
      v8bf u3 = *(const v8bf*)(pBl + 16);
      v16bf Blo = __builtin_shufflevector(u2, u3, 0,1,2,3,4,5,6,7,8,9,10,11,12,13,14,15);

      acc0[nt] = __builtin_amdgcn_wmma_f32_16x16x32_bf16(
          false, Ahi0, false, Bhi, (short)0, acc0[nt], false, false);
      acc0[nt] = __builtin_amdgcn_wmma_f32_16x16x32_bf16(
          false, Ahi0, false, Blo, (short)0, acc0[nt], false, false);
      acc0[nt] = __builtin_amdgcn_wmma_f32_16x16x32_bf16(
          false, Alo0, false, Bhi, (short)0, acc0[nt], false, false);

      acc1[nt] = __builtin_amdgcn_wmma_f32_16x16x32_bf16(
          false, Ahi1, false, Bhi, (short)0, acc1[nt], false, false);
      acc1[nt] = __builtin_amdgcn_wmma_f32_16x16x32_bf16(
          false, Ahi1, false, Blo, (short)0, acc1[nt], false, false);
      acc1[nt] = __builtin_amdgcn_wmma_f32_16x16x32_bf16(
          false, Alo1, false, Bhi, (short)0, acc1[nt], false, false);
    }
  }

  // epilogue: + b1, relu, store f32
#pragma unroll
  for (int nt = 0; nt < 4; ++nt) {
    const int col = nbase + 16 * nt + l16;
    const float bias = b1[col];
#pragma unroll
    for (int r = 0; r < 8; ++r) {
      const int row0 = m0 + r + 8 * half;
      h1[(size_t)row0 * H_ + col] = fmaxf(acc0[nt][r] + bias, 0.f);
      h1[(size_t)(row0 + 16) * H_ + col] = fmaxf(acc1[nt][r] + bias, 0.f);
    }
  }
}

// =====================================================================
// logits = H1 @ w2 + b2 + adj[b]   (N=8 reduction kernel, 1 block/token)
// =====================================================================
__global__ __launch_bounds__(256)
void logits_kernel(const float* __restrict__ h1, const float* __restrict__ w2,
                   const float* __restrict__ b2, const float* __restrict__ adj,
                   float* __restrict__ logits) {
  __shared__ float red[256][E_];
  const int t = blockIdx.x, tid = threadIdx.x;
  float acc[E_] = {0, 0, 0, 0, 0, 0, 0, 0};
  for (int k = tid; k < H_; k += 256) {
    float x = h1[(size_t)t * H_ + k];
    const float* w = w2 + (size_t)k * E_;
#pragma unroll
    for (int e = 0; e < E_; ++e) acc[e] += x * w[e];
  }
#pragma unroll
  for (int e = 0; e < E_; ++e) red[tid][e] = acc[e];
  __syncthreads();
  for (int s = 128; s > 0; s >>= 1) {
    if (tid < s) {
#pragma unroll
      for (int e = 0; e < E_; ++e) red[tid][e] += red[tid + s][e];
    }
    __syncthreads();
  }
  if (tid < E_) {
    int b = t / S_;
    logits[(size_t)t * E_ + tid] = red[0][tid] + b2[tid] + adj[b * E_ + tid];
  }
}

// =====================================================================
// softmax over E=8, top-2 (jax tie-break: lowest index), renorm,
// + accumulate aux-loss statistics.
// =====================================================================
__global__ __launch_bounds__(256)
void topk_kernel(const float* __restrict__ logits, float* __restrict__ probs_out,
                 int* __restrict__ e_flat, float* __restrict__ p_flat,
                 float* __restrict__ psum, unsigned int* __restrict__ cnt) {
  __shared__ float s_psum[E_];
  __shared__ unsigned int s_cnt[E_];
  if (threadIdx.x < E_) { s_psum[threadIdx.x] = 0.f; s_cnt[threadIdx.x] = 0u; }
  __syncthreads();

  const int t = blockIdx.x * 256 + threadIdx.x;
  if (t < MTOT) {
    float l[E_], mx = -1e30f;
#pragma unroll
    for (int e = 0; e < E_; ++e) { l[e] = logits[(size_t)t * E_ + e]; mx = fmaxf(mx, l[e]); }
    float sum = 0.f;
#pragma unroll
    for (int e = 0; e < E_; ++e) { l[e] = expf(l[e] - mx); sum += l[e]; }
    const float inv = 1.f / sum;
    float p[E_];
#pragma unroll
    for (int e = 0; e < E_; ++e) {
      p[e] = l[e] * inv;
      probs_out[(size_t)t * E_ + e] = p[e];
      atomicAdd(&s_psum[e], p[e]);
    }
    int e1 = 0; float p1 = p[0];
#pragma unroll
    for (int e = 1; e < E_; ++e) if (p[e] > p1) { p1 = p[e]; e1 = e; }
    int e2 = -1; float p2 = -1e30f;
#pragma unroll
    for (int e = 0; e < E_; ++e) if (e != e1 && p[e] > p2) { p2 = p[e]; e2 = e; }
    const float inv2 = 1.f / (p1 + p2 + 1e-8f);
    e_flat[2 * t]     = e1;  p_flat[2 * t]     = p1 * inv2;
    e_flat[2 * t + 1] = e2;  p_flat[2 * t + 1] = p2 * inv2;
    atomicAdd(&s_cnt[e1], 1u);
    atomicAdd(&s_cnt[e2], 1u);
  }
  __syncthreads();
  if (threadIdx.x < E_) {
    atomicAdd(&psum[threadIdx.x], s_psum[threadIdx.x]);
    atomicAdd(&cnt[threadIdx.x], s_cnt[threadIdx.x]);
  }
}

// =====================================================================
// Serial-order capacity assignment: blocked exclusive scan over 8192
// slots with 8 per-expert counters, then scatter into dispatch/combine.
// Single workgroup (256 threads x 32 slots).
// =====================================================================
__global__ __launch_bounds__(256)
void scatter_kernel(const int* __restrict__ e_flat, const float* __restrict__ p_flat,
                    float* __restrict__ dispatch, float* __restrict__ combine) {
  __shared__ int sc[256][E_];
  const int tid = threadIdx.x;
  const int per = NSLOT / 256;   // 32
  const int base = tid * per;

  int local[E_] = {0, 0, 0, 0, 0, 0, 0, 0};
  for (int i = 0; i < per; ++i) local[e_flat[base + i]]++;
#pragma unroll
  for (int e = 0; e < E_; ++e) sc[tid][e] = local[e];
  __syncthreads();

  if (tid < E_) {              // exclusive scan per expert across threads
    int run = 0;
    for (int t = 0; t < 256; ++t) { int v = sc[t][tid]; sc[t][tid] = run; run += v; }
  }
  __syncthreads();

  int ctr[E_];
#pragma unroll
  for (int e = 0; e < E_; ++e) ctr[e] = sc[tid][e];

  for (int i = 0; i < per; ++i) {
    const int idx = base + i;
    const int e = e_flat[idx];
    const int pos = ctr[e]++;
    if (pos < CAP) {
      const int tok = idx >> 1;
      const size_t off = (size_t)tok * (E_ * CAP) + (size_t)e * CAP + pos;
      dispatch[off] = 1.0f;
      combine[off]  = p_flat[idx];
    }
  }
}

// =====================================================================
// aux_loss = E * sum_e mean_prob[e] * usage[e]
// =====================================================================
__global__ void aux_kernel(const float* __restrict__ psum,
                           const unsigned int* __restrict__ cnt,
                           float* __restrict__ out_aux) {
  if (threadIdx.x == 0 && blockIdx.x == 0) {
    float a = 0.f;
    for (int e = 0; e < E_; ++e)
      a += (psum[e] / (float)MTOT) * ((float)cnt[e] / (float)NSLOT);
    *out_aux = a * (float)E_;
  }
}

// =====================================================================
// launch
// =====================================================================
extern "C" void kernel_launch(void* const* d_in, const int* in_sizes, int n_in,
                              void* d_out, int out_size, void* d_ws, size_t ws_size,
                              hipStream_t stream) {
  (void)in_sizes; (void)n_in; (void)out_size; (void)ws_size;
  const float* hs  = (const float*)d_in[0];
  const float* w1  = (const float*)d_in[1];
  const float* b1  = (const float*)d_in[2];
  const float* w2  = (const float*)d_in[3];
  const float* b2  = (const float*)d_in[4];
  const float* wc1 = (const float*)d_in[5];
  const float* bc1 = (const float*)d_in[6];
  const float* wc2 = (const float*)d_in[7];
  const float* bc2 = (const float*)d_in[8];
  const float* chr = (const float*)d_in[9];

  float* out = (float*)d_out;
  float* dispatch = out;
  float* combine  = out + DISPSZ;
  float* probs    = out + 2 * DISPSZ;
  float* aux      = out + 2 * DISPSZ + (size_t)MTOT * E_;

  char* ws = (char*)d_ws;
  __bf16* xhi   = (__bf16*)(ws + OFF_XHI);
  __bf16* xlo   = (__bf16*)(ws + OFF_XLO);
  __bf16* wthi  = (__bf16*)(ws + OFF_W1THI);
  __bf16* wtlo  = (__bf16*)(ws + OFF_W1TLO);
  float* h1     = (float*)(ws + OFF_H1);
  float* logits = (float*)(ws + OFF_LOGITS);
  float* avg    = (float*)(ws + OFF_AVG);
  float* adj    = (float*)(ws + OFF_ADJ);
  int*   eflat  = (int*)(ws + OFF_EFLAT);
  float* pflat  = (float*)(ws + OFF_PFLAT);
  float* psum   = (float*)(ws + OFF_PSUM);
  unsigned int* cnt = (unsigned int*)(ws + OFF_CNT);

  // zero the sparse outputs and the aux accumulators (graph-capturable)
  hipMemsetAsync(dispatch, 0, 2 * DISPSZ * sizeof(float), stream);
  hipMemsetAsync(psum, 0, 64 + 64, stream);   // psum[8] + cnt[8] regions

  // precision split / transpose
  split_x_kernel<<<(MTOT * H_ + 255) / 256, 256, 0, stream>>>(hs, xhi, xlo, MTOT * H_);
  split_w1t_kernel<<<(H_ * H_ + 255) / 256, 256, 0, stream>>>(w1, wthi, wtlo);

  // cache-aware adjustment path
  avg_kernel<<<dim3(H_ / 256, B_), 256, 0, stream>>>(hs, avg);
  cacheadj_kernel<<<B_, 256, 0, stream>>>(avg, chr, wc1, bc1, wc2, bc2, adj);

  // main WMMA GEMM: 128 threads (4 waves), WG tile 128x64
  gemm1_wmma_kernel<<<dim3(MTOT / 128, H_ / 64), 128, 0, stream>>>(
      xhi, xlo, wthi, wtlo, b1, h1);

  // router logits, softmax/top-2, scatter, aux loss
  logits_kernel<<<MTOT, 256, 0, stream>>>(h1, w2, b2, adj, logits);
  topk_kernel<<<MTOT / 256, 256, 0, stream>>>(logits, probs, eflat, pflat, psum, cnt);
  scatter_kernel<<<1, 256, 0, stream>>>(eflat, pflat, dispatch, combine);
  aux_kernel<<<1, 32, 0, stream>>>(psum, cnt, aux);
}
